// LinearTransformerEncoder_33930241639084
// MI455X (gfx1250) — matrix operand order repaired and checked by reading, not compiled, not run on hardware
//
#include <hip/hip_runtime.h>
#include <hip/hip_bf16.h>
#include <hip/hip_fp16.h>

typedef __attribute__((ext_vector_type(16))) _Float16 v16h;
typedef __attribute__((ext_vector_type(8)))  float    v8f;
typedef __attribute__((ext_vector_type(4)))  unsigned v4u;

namespace {

constexpr int Bz  = 2;
constexpr int Sz  = 1024;
constexpr int Lz  = 2;
constexpr int Hz  = 8;
constexpr int Dz  = 512;
constexpr int DFz = 2048;
constexpr int Mz  = Bz * Sz;        // 2048 rows (tokens)
constexpr int DK  = 64;             // head dim
constexpr int CHUNK  = 128;
constexpr int NCHUNK = Sz / CHUNK;  // 8

constexpr int TM = 128;             // block tile M
constexpr int TN = 64;              // block tile N
constexpr int TK = 32;              // K step
constexpr int SAS = 40;             // sA stride in halves (mult of 8)
constexpr int SBS = 40;             // sBt stride in halves (mult of 8)

union F16x16 { v16h v; _Float16 h[16]; };
union F32x8  { v8f  v; float    f[8];  };

} // namespace

// ---------------------------------------------------------------------------
// Elementwise: f32 -> (optional f32 copy) + f16 cast
// ---------------------------------------------------------------------------
__global__ void lte_copycast_kernel(const float* __restrict__ in,
                                    float* __restrict__ outF,
                                    _Float16* __restrict__ outH, int n) {
  int i = blockIdx.x * blockDim.x + threadIdx.x;
  if (i < n) {
    float v = in[i];
    if (outF) outF[i] = v;
    outH[i] = (_Float16)v;
  }
}

// ---------------------------------------------------------------------------
// WMMA GEMM: out[M,N] = A[M,K] (f16) x Bw[K,N] (f16) + bias[N]
// Block: 256 thr = 8 waves; block tile 128x64; K-step 32 via LDS.
// Wave w: rows (w&3)*32, cols (w>>2)*32 -> 2x2 = 4 WMMA 16x16 tiles.
// A staged row-major (stride 40 halves); B staged TRANSPOSED [col][k]
// (stride 40 halves) so every fragment is two contiguous ds_load_b128.
// ---------------------------------------------------------------------------
__global__ __launch_bounds__(256)
void lte_wmma_gemm_kernel(const _Float16* __restrict__ A,
                          const _Float16* __restrict__ Bw,
                          const float* __restrict__ bias,
                          float* __restrict__ outF,
                          _Float16* __restrict__ outH,
                          int M, int N, int K, int relu) {
  __shared__ alignas(16) _Float16 sA[TM * SAS];   // 128 rows x 32 K
  __shared__ alignas(16) _Float16 sBt[TN * SBS];  // 64 cols x 32 K (transposed)

  const int tid  = threadIdx.x;
  const int wave = tid >> 5;
  const int lane = tid & 31;
  const int m0 = blockIdx.y * TM;
  const int n0 = blockIdx.x * TN;
  const int mrow = (wave & 3) * 32;   // wave's 32-row strip
  const int ncol = (wave >> 2) * 32;  // wave's 32-col strip

  const int lr    = lane & 15;
  const int kaddA = (lane >= 16) ? 8 : 0;    // A frag K base (halves)
  const int kaddB = (lane >= 16) ? 8 : 0;    // B frag K base (halves)

  // Staging coordinates (fixed per thread).
  const int act = tid & 127;          // A row within tile (id & 127)
  const int bct = tid & 63;           // B column within tile
  const int bkb = (tid >> 6) * 8;     // B k-chunk base (0,8,16,24)

  v8f acc[2][2];
#pragma unroll
  for (int mi = 0; mi < 2; ++mi)
#pragma unroll
    for (int ni = 0; ni < 2; ++ni)
      acc[mi][ni] = (v8f){0.f, 0.f, 0.f, 0.f, 0.f, 0.f, 0.f, 0.f};

  for (int kk = 0; kk < K; kk += TK) {
    // ---- Stage A: 128x32 halves, one b128 global load + b128 LDS store x2.
#pragma unroll
    for (int i = 0; i < 2; ++i) {
      int id = tid + i * 256;          // 0..511
      int r  = id & 127;
      int cb = (id >> 7) * 8;          // 0,8,16,24
      v4u d = *reinterpret_cast<const v4u*>(
          A + (size_t)(m0 + r) * K + kk + cb);
      *reinterpret_cast<v4u*>(&sA[r * SAS + cb]) = d;
    }
    // ---- Stage B transposed: thread owns (col=bct, k=bkb..bkb+7).
    {
      alignas(16) _Float16 tmp[8];
#pragma unroll
      for (int j = 0; j < 8; ++j)
        tmp[j] = Bw[(size_t)(kk + bkb + j) * N + n0 + bct];
      *reinterpret_cast<v4u*>(&sBt[bct * SBS + bkb]) =
          *reinterpret_cast<const v4u*>(tmp);
    }
    // Prefetch next K tile while this one is consumed.
    if (kk + TK < K) {
      __builtin_prefetch(A + (size_t)(m0 + act) * K + kk + TK, 0, 1);
      __builtin_prefetch(Bw + (size_t)(kk + TK + bkb) * N + n0 + bct, 0, 1);
    }
    __syncthreads();

    // ---- Fragments: each = two contiguous 16B LDS loads.
    F16x16 fa[2], fb[2];
#pragma unroll
    for (int mi = 0; mi < 2; ++mi) {
      const _Float16* base = &sA[(mrow + mi * 16 + lr) * SAS + kaddA];
      *reinterpret_cast<v4u*>(&fa[mi].h[0]) =
          *reinterpret_cast<const v4u*>(base);
      *reinterpret_cast<v4u*>(&fa[mi].h[8]) =
          *reinterpret_cast<const v4u*>(base + 16);
    }
#pragma unroll
    for (int ni = 0; ni < 2; ++ni) {
      const _Float16* base = &sBt[(ncol + ni * 16 + lr) * SBS + kaddB];
      *reinterpret_cast<v4u*>(&fb[ni].h[0]) =
          *reinterpret_cast<const v4u*>(base);
      *reinterpret_cast<v4u*>(&fb[ni].h[8]) =
          *reinterpret_cast<const v4u*>(base + 16);
    }

#pragma unroll
    for (int mi = 0; mi < 2; ++mi)
#pragma unroll
      for (int ni = 0; ni < 2; ++ni)
        acc[mi][ni] = __builtin_amdgcn_wmma_f32_16x16x32_f16(
            false, fa[mi].v, false, fb[ni].v, (short)0, acc[mi][ni], false,
            false);
    __syncthreads();
  }

  // ---- Epilogue: C/D layout — VGPR i: lanes 0-15 -> M=i, lanes 16-31 -> M=8+i.
  const int rofs = (lane >= 16) ? 8 : 0;
#pragma unroll
  for (int mi = 0; mi < 2; ++mi) {
#pragma unroll
    for (int ni = 0; ni < 2; ++ni) {
      F32x8 r;
      r.v = acc[mi][ni];
      const int col = n0 + ncol + ni * 16 + lr;
      const float bcol = bias[col];
#pragma unroll
      for (int i = 0; i < 8; ++i) {
        int row = m0 + mrow + mi * 16 + rofs + i;
        float v = r.f[i] + bcol;
        if (relu) v = fmaxf(v, 0.f);
        size_t o = (size_t)row * N + col;
        if (outF) outF[o] = v;
        if (outH) outH[o] = (_Float16)v;
      }
    }
  }
}

// ---------------------------------------------------------------------------
// Linear attention, chunk-parallel scan over S.
// Thread t owns value-column v = t of the 64x64 KV state (64 regs).
// ---------------------------------------------------------------------------
__global__ __launch_bounds__(64)
void lte_attn_chunk_kv_kernel(const float* __restrict__ k,
                              const float* __restrict__ v,
                              float* __restrict__ kvbuf) {
  const int c  = blockIdx.x;          // chunk
  const int bh = blockIdx.y;          // b*H + h
  const int b  = bh / Hz;
  const int h  = bh % Hz;
  const int t  = threadIdx.x;
  __shared__ float lk[DK];

  float st[DK];
#pragma unroll
  for (int i = 0; i < DK; ++i) st[i] = 0.f;

  const int s0 = c * CHUNK;
  for (int s = 0; s < CHUNK; ++s) {
    size_t base = ((size_t)b * Sz + (s0 + s)) * Dz + h * DK;
    lk[t] = k[base + t];
    float vv = v[base + t];
    __syncthreads();
#pragma unroll
    for (int i = 0; i < DK; ++i) st[i] = fmaf(lk[i], vv, st[i]);
    __syncthreads();
  }
  size_t ob = ((size_t)bh * NCHUNK + c) * (DK * DK);
#pragma unroll
  for (int i = 0; i < DK; ++i) kvbuf[ob + (size_t)i * DK + t] = st[i];
}

__global__ __launch_bounds__(64)
void lte_attn_prefix_kernel(float* __restrict__ kvbuf) {
  const int bh = blockIdx.x;
  const int t  = threadIdx.x;
  float acc[DK];
#pragma unroll
  for (int i = 0; i < DK; ++i) acc[i] = 0.f;
  for (int c = 0; c < NCHUNK; ++c) {
    size_t ob = ((size_t)bh * NCHUNK + c) * (DK * DK);
#pragma unroll
    for (int i = 0; i < DK; ++i) {
      size_t idx = ob + (size_t)i * DK + t;
      float tmp = kvbuf[idx];
      kvbuf[idx] = acc[i];     // exclusive prefix
      acc[i] += tmp;
    }
  }
}

__global__ __launch_bounds__(64)
void lte_attn_scan_kernel(const float* __restrict__ q,
                          const float* __restrict__ k,
                          const float* __restrict__ v,
                          const float* __restrict__ kvbuf,
                          _Float16* __restrict__ atth) {
  const int c  = blockIdx.x;
  const int bh = blockIdx.y;
  const int b  = bh / Hz;
  const int h  = bh % Hz;
  const int t  = threadIdx.x;
  __shared__ float lk[DK];
  __shared__ float lq[DK];

  float st[DK];
  size_t ob = ((size_t)bh * NCHUNK + c) * (DK * DK);
#pragma unroll
  for (int i = 0; i < DK; ++i) st[i] = kvbuf[ob + (size_t)i * DK + t];

  const int s0 = c * CHUNK;
  for (int s = 0; s < CHUNK; ++s) {
    size_t base = ((size_t)b * Sz + (s0 + s)) * Dz + h * DK;
    lk[t] = k[base + t];
    lq[t] = q[base + t];
    float vv = v[base + t];
    __syncthreads();
    float o = 0.f;
#pragma unroll
    for (int i = 0; i < DK; ++i) {
      st[i] = fmaf(lk[i], vv, st[i]);   // inclusive cumsum of k (x) v
      o = fmaf(lq[i], st[i], o);        // q . state
    }
    atth[base + t] = (_Float16)o;
    __syncthreads();
  }
}

// ---------------------------------------------------------------------------
// Fused residual + LayerNorm: out = LN(x + y; g, b); optional f16 emit.
// One block per row, D=512, 256 threads (2 elems each).
// ---------------------------------------------------------------------------
__global__ __launch_bounds__(256)
void lte_add_ln_kernel(const float* __restrict__ xr,
                       const float* __restrict__ y,
                       const float* __restrict__ g,
                       const float* __restrict__ bb,
                       float* __restrict__ outF,
                       _Float16* __restrict__ outH) {
  const int row = blockIdx.x;
  const int t   = threadIdx.x;
  __shared__ float red[256];

  size_t base = (size_t)row * Dz;
  float v0 = xr[base + t]       + (y ? y[base + t]       : 0.f);
  float v1 = xr[base + t + 256] + (y ? y[base + t + 256] : 0.f);

  red[t] = v0 + v1;
  __syncthreads();
  for (int off = 128; off > 0; off >>= 1) {
    if (t < off) red[t] += red[t + off];
    __syncthreads();
  }
  float mean = red[0] * (1.0f / Dz);
  __syncthreads();

  float d0 = v0 - mean, d1 = v1 - mean;
  red[t] = d0 * d0 + d1 * d1;
  __syncthreads();
  for (int off = 128; off > 0; off >>= 1) {
    if (t < off) red[t] += red[t + off];
    __syncthreads();
  }
  float inv = rsqrtf(red[0] * (1.0f / Dz) + 1e-5f);

  float o0 = d0 * inv * g[t]       + bb[t];
  float o1 = d1 * inv * g[t + 256] + bb[t + 256];
  if (outF) {
    outF[base + t] = o0;
    outF[base + t + 256] = o1;
  }
  if (outH) {
    outH[base + t] = (_Float16)o0;
    outH[base + t + 256] = (_Float16)o1;
  }
}

// ---------------------------------------------------------------------------
// Host-side orchestration
// ---------------------------------------------------------------------------
static inline void lte_gemm(const _Float16* A, const _Float16* Bw,
                            const float* bias, float* outF, _Float16* outH,
                            int M, int N, int K, int relu, hipStream_t s) {
  dim3 grid(N / TN, M / TM), block(256);
  lte_wmma_gemm_kernel<<<grid, block, 0, s>>>(A, Bw, bias, outF, outH, M, N, K,
                                              relu);
}

static inline void lte_cast(const float* in, float* outF, _Float16* outH, int n,
                            hipStream_t s) {
  lte_copycast_kernel<<<(n + 255) / 256, 256, 0, s>>>(in, outF, outH, n);
}

extern "C" void kernel_launch(void* const* d_in, const int* in_sizes, int n_in,
                              void* d_out, int out_size, void* d_ws,
                              size_t ws_size, hipStream_t stream) {
  (void)in_sizes; (void)n_in; (void)out_size; (void)ws_size;

  const float* src = (const float*)d_in[0];
  const float* Wq  = (const float*)d_in[1];
  const float* bq  = (const float*)d_in[2];
  const float* Wk  = (const float*)d_in[3];
  const float* bk  = (const float*)d_in[4];
  const float* Wv  = (const float*)d_in[5];
  const float* bv  = (const float*)d_in[6];
  const float* Wo  = (const float*)d_in[7];
  const float* bo  = (const float*)d_in[8];
  const float* W1  = (const float*)d_in[9];
  const float* b1  = (const float*)d_in[10];
  const float* W2  = (const float*)d_in[11];
  const float* b2  = (const float*)d_in[12];
  const float* g1  = (const float*)d_in[13];
  const float* be1 = (const float*)d_in[14];
  const float* g2  = (const float*)d_in[15];
  const float* be2 = (const float*)d_in[16];
  const float* gf  = (const float*)d_in[17];
  const float* bef = (const float*)d_in[18];

  // Workspace carve-out (256B aligned slices).
  char* p = (char*)d_ws;
  auto alloc = [&](size_t bytes) -> void* {
    void* r = (void*)p;
    p += (bytes + 255) & ~(size_t)255;
    return r;
  };
  float*    xf32  = (float*)   alloc((size_t)Mz * Dz * 4);
  _Float16* xh    = (_Float16*)alloc((size_t)Mz * Dz * 2);
  float*    qf    = (float*)   alloc((size_t)Mz * Dz * 4);
  float*    kf    = (float*)   alloc((size_t)Mz * Dz * 4);
  float*    vf    = (float*)   alloc((size_t)Mz * Dz * 4);
  float*    of    = (float*)   alloc((size_t)Mz * Dz * 4);
  _Float16* atth  = (_Float16*)alloc((size_t)Mz * Dz * 2);
  _Float16* ff1h  = (_Float16*)alloc((size_t)Mz * DFz * 2);
  float*    kvbuf = (float*)   alloc((size_t)Bz * Hz * NCHUNK * DK * DK * 4);
  _Float16* wqh   = (_Float16*)alloc((size_t)Lz * Dz * Dz * 2);
  _Float16* wkh   = (_Float16*)alloc((size_t)Lz * Dz * Dz * 2);
  _Float16* wvh   = (_Float16*)alloc((size_t)Lz * Dz * Dz * 2);
  _Float16* woh   = (_Float16*)alloc((size_t)Lz * Dz * Dz * 2);
  _Float16* w1h   = (_Float16*)alloc((size_t)Lz * Dz * DFz * 2);
  _Float16* w2h   = (_Float16*)alloc((size_t)Lz * DFz * Dz * 2);

  // x = src (f32 copy) + f16 working copy.
  lte_cast(src, xf32, xh, Mz * Dz, stream);
  // Weights -> f16 (every call; deterministic, no caching).
  lte_cast(Wq, nullptr, wqh, Lz * Dz * Dz, stream);
  lte_cast(Wk, nullptr, wkh, Lz * Dz * Dz, stream);
  lte_cast(Wv, nullptr, wvh, Lz * Dz * Dz, stream);
  lte_cast(Wo, nullptr, woh, Lz * Dz * Dz, stream);
  lte_cast(W1, nullptr, w1h, Lz * Dz * DFz, stream);
  lte_cast(W2, nullptr, w2h, Lz * DFz * Dz, stream);

  for (int i = 0; i < Lz; ++i) {
    const size_t wdd = (size_t)i * Dz * Dz;
    const size_t wdf = (size_t)i * Dz * DFz;
    // Q / K / V projections (f32 out for the scan).
    lte_gemm(xh, wqh + wdd, bq + i * Dz, qf, nullptr, Mz, Dz, Dz, 0, stream);
    lte_gemm(xh, wkh + wdd, bk + i * Dz, kf, nullptr, Mz, Dz, Dz, 0, stream);
    lte_gemm(xh, wvh + wdd, bv + i * Dz, vf, nullptr, Mz, Dz, Dz, 0, stream);
    // Causal linear attention: chunk KV sums -> exclusive prefix -> scan.
    lte_attn_chunk_kv_kernel<<<dim3(NCHUNK, Bz * Hz), 64, 0, stream>>>(kf, vf,
                                                                       kvbuf);
    lte_attn_prefix_kernel<<<Bz * Hz, 64, 0, stream>>>(kvbuf);
    lte_attn_scan_kernel<<<dim3(NCHUNK, Bz * Hz), 64, 0, stream>>>(
        qf, kf, vf, kvbuf, atth);
    // Output projection.
    lte_gemm(atth, woh + wdd, bo + i * Dz, of, nullptr, Mz, Dz, Dz, 0, stream);
    // x = LN(x + att)  (writes f32 + f16 copies).
    lte_add_ln_kernel<<<Mz, 256, 0, stream>>>(xf32, of, g1 + i * Dz,
                                              be1 + i * Dz, xf32, xh);
    // FFN: relu(x @ W1 + b1) -> f16 only; then @ W2 + b2 -> f32.
    lte_gemm(xh, w1h + wdf, b1 + i * DFz, nullptr, ff1h, Mz, DFz, Dz, 1,
             stream);
    lte_gemm(ff1h, w2h + wdf, b2 + i * Dz, of, nullptr, Mz, Dz, DFz, 0,
             stream);
    // x = LN(x + ff).
    lte_add_ln_kernel<<<Mz, 256, 0, stream>>>(xf32, of, g2 + i * Dz,
                                              be2 + i * Dz, xf32, xh);
  }
  // Final LayerNorm -> d_out (f32).
  lte_add_ln_kernel<<<Mz, 256, 0, stream>>>(xf32, nullptr, gf, bef,
                                            (float*)d_out, nullptr);
}
